// TransformerBlock_59459527246572
// MI455X (gfx1250) — matrix operand order repaired and checked by reading, not compiled, not run on hardware
//
#include <hip/hip_runtime.h>

#define Bv   2
#define Lv   512
#define Dv   128
#define Pv   64
#define Hv   4
#define HDv  32
#define FFNv 512

typedef _Float16 h8  __attribute__((ext_vector_type(8)));
typedef _Float16 h16 __attribute__((ext_vector_type(16)));
typedef float    v8f __attribute__((ext_vector_type(8)));

__device__ __forceinline__ void wavesync() {
    __builtin_amdgcn_wave_barrier();
    asm volatile("s_wait_dscnt 0" ::: "memory");
    __builtin_amdgcn_wave_barrier();
}

// ---------------------------------------------------------------------------
// Weight prep: f32 -> f16, laid out exactly as the pair kernel's LDS wants.
//  wp16[h*64+p] = pmlp_w1[h, 256+p]   (Wp block, row-major [h][p])
//  w216[n*128+k] = pmlp_w2[n, k]      (row-major [n][k])
// ---------------------------------------------------------------------------
__global__ void wprep_kernel(const float* __restrict__ pmlp_w1,
                             const float* __restrict__ pmlp_w2,
                             _Float16* __restrict__ wp16,
                             _Float16* __restrict__ w216) {
    int i = blockIdx.x * blockDim.x + threadIdx.x;
    if (i < 128 * 64) {
        int h = i >> 6, p = i & 63;
        wp16[i] = (_Float16)pmlp_w1[h * 320 + 256 + p];
        w216[i] = (_Float16)pmlp_w2[i];
    }
}

// ---------------------------------------------------------------------------
// LN1 + QKV projection. One block (128 thr) per token row.
// ---------------------------------------------------------------------------
__global__ void ln_qkv_kernel(const float* __restrict__ x_in,
                              const float* __restrict__ ln_w,
                              const float* __restrict__ ln_b,
                              const float* __restrict__ w,
                              const float* __restrict__ bias,
                              float* __restrict__ qkv) {
    int row = blockIdx.x;
    int tid = threadIdx.x; // 128
    __shared__ float sx[Dv];
    __shared__ float red[Dv];
    float xv = x_in[row * Dv + tid];
    red[tid] = xv; __syncthreads();
    for (int s = 64; s > 0; s >>= 1) { if (tid < s) red[tid] += red[tid + s]; __syncthreads(); }
    float mean = red[0] * (1.f / Dv); __syncthreads();
    float d = xv - mean;
    red[tid] = d * d; __syncthreads();
    for (int s = 64; s > 0; s >>= 1) { if (tid < s) red[tid] += red[tid + s]; __syncthreads(); }
    float rstd = rsqrtf(red[0] * (1.f / Dv) + 1e-5f);
    sx[tid] = d * rstd * ln_w[tid] + ln_b[tid];
    __syncthreads();
    for (int s = 0; s < 3; ++s) {
        int o = tid + s * Dv;
        float acc = bias[o];
        const float* wr = w + (size_t)o * Dv;
        for (int e = 0; e < Dv; ++e) acc += sx[e] * wr[e];
        qkv[(size_t)row * (3 * Dv) + o] = acc;
    }
}

// ---------------------------------------------------------------------------
// Attention: one block (128 thr) per (b, h, l). Softmax over m=0..511.
// ---------------------------------------------------------------------------
__global__ void attn_kernel(const float* __restrict__ qkv,
                            const unsigned char* __restrict__ mask,
                            float* __restrict__ ctx) {
    int blk = blockIdx.x;
    int b = blk / (Hv * Lv);
    int rem = blk % (Hv * Lv);
    int h = rem / Lv;
    int l = rem % Lv;
    int tid = threadIdx.x; // 128
    __shared__ float sq[HDv];
    __shared__ float sc[Lv];
    __shared__ float red[128];
    if (tid < HDv) sq[tid] = qkv[(size_t)(b * Lv + l) * (3 * Dv) + h * HDv + tid];
    __syncthreads();
    float lmax = -3.0e38f;
    for (int m = tid; m < Lv; m += 128) {
        const float* kp = qkv + (size_t)(b * Lv + m) * (3 * Dv) + Dv + h * HDv;
        float s = 0.f;
        for (int d = 0; d < HDv; ++d) s += sq[d] * kp[d];
        s *= 0.17677669529663687f; // 1/sqrt(32)
        if (!mask[b * Lv + m]) s = -1e9f;
        sc[m] = s;
        lmax = fmaxf(lmax, s);
    }
    red[tid] = lmax; __syncthreads();
    for (int s = 64; s > 0; s >>= 1) { if (tid < s) red[tid] = fmaxf(red[tid], red[tid + s]); __syncthreads(); }
    float gmax = red[0]; __syncthreads();
    float lsum = 0.f;
    for (int m = tid; m < Lv; m += 128) {
        float p = __expf(sc[m] - gmax);
        sc[m] = p; lsum += p;
    }
    red[tid] = lsum; __syncthreads();
    for (int s = 64; s > 0; s >>= 1) { if (tid < s) red[tid] += red[tid + s]; __syncthreads(); }
    float inv = 1.f / red[0];
    __syncthreads();
    int d = tid & (HDv - 1);
    int seg = tid >> 5;
    float part = 0.f;
    for (int m = seg * 128; m < seg * 128 + 128; ++m)
        part += sc[m] * qkv[(size_t)(b * Lv + m) * (3 * Dv) + 2 * Dv + h * HDv + d];
    red[tid] = part; __syncthreads();
    if (tid < HDv) {
        float c = (red[d] + red[d + 32] + red[d + 64] + red[d + 96]) * inv;
        ctx[(size_t)(b * Lv + l) * Dv + h * HDv + d] = c;
    }
}

// ---------------------------------------------------------------------------
// out_proj + residual + LN2 + FFN + residual + mask. Block (256 thr) per row.
// ---------------------------------------------------------------------------
__global__ void outffn_kernel(const float* __restrict__ residue_in,
                              const float* __restrict__ ctx,
                              const float* __restrict__ opw, const float* __restrict__ opb,
                              const float* __restrict__ ln2w, const float* __restrict__ ln2b,
                              const float* __restrict__ w1, const float* __restrict__ b1,
                              const float* __restrict__ w2, const float* __restrict__ b2,
                              const unsigned char* __restrict__ mask,
                              float* __restrict__ res_out) {
    int row = blockIdx.x;
    int tid = threadIdx.x; // 256
    __shared__ float sc[Dv];
    __shared__ float sy[Dv];
    __shared__ float sh[FFNv];
    __shared__ float red[256];
    if (tid < Dv) sc[tid] = ctx[(size_t)row * Dv + tid];
    __syncthreads();
    float r1 = 0.f;
    if (tid < Dv) {
        float a = opb[tid];
        const float* wr = opw + (size_t)tid * Dv;
        for (int e = 0; e < Dv; ++e) a += sc[e] * wr[e];
        r1 = residue_in[(size_t)row * Dv + tid] + a;
    }
    red[tid] = (tid < Dv) ? r1 : 0.f; __syncthreads();
    for (int s = 128; s > 0; s >>= 1) { if (tid < s) red[tid] += red[tid + s]; __syncthreads(); }
    float mean = red[0] * (1.f / Dv); __syncthreads();
    float dd = r1 - mean;
    red[tid] = (tid < Dv) ? dd * dd : 0.f; __syncthreads();
    for (int s = 128; s > 0; s >>= 1) { if (tid < s) red[tid] += red[tid + s]; __syncthreads(); }
    float rstd = rsqrtf(red[0] * (1.f / Dv) + 1e-5f);
    if (tid < Dv) sy[tid] = dd * rstd * ln2w[tid] + ln2b[tid];
    __syncthreads();
    for (int f = tid; f < FFNv; f += 256) {
        float a = b1[f];
        const float* wr = w1 + (size_t)f * Dv;
        for (int e = 0; e < Dv; ++e) a += sy[e] * wr[e];
        sh[f] = fmaxf(a, 0.f);
    }
    __syncthreads();
    if (tid < Dv) {
        float a = b2[tid];
        const float* wr = w2 + (size_t)tid * FFNv;
        for (int f = 0; f < FFNv; ++f) a += sh[f] * wr[f];
        float mv = mask[row] ? 1.f : 0.f;
        res_out[(size_t)row * Dv + tid] = (r1 + a) * mv;
    }
}

// ---------------------------------------------------------------------------
// hi = residue @ Wi.T, hj = residue @ Wj.T. Block (256 thr) per row.
// ---------------------------------------------------------------------------
__global__ void hij_kernel(const float* __restrict__ res,
                           const float* __restrict__ pmlp_w1,
                           float* __restrict__ hi, float* __restrict__ hj) {
    int row = blockIdx.x;
    int tid = threadIdx.x; // 256
    __shared__ float sr[Dv];
    if (tid < Dv) sr[tid] = res[(size_t)row * Dv + tid];
    __syncthreads();
    if (tid < 128) {
        float a = 0.f;
        const float* wr = pmlp_w1 + (size_t)tid * 320;
        for (int e = 0; e < Dv; ++e) a += sr[e] * wr[e];
        hi[(size_t)row * 128 + tid] = a;
    } else {
        int t = tid - 128;
        float a = 0.f;
        const float* wr = pmlp_w1 + (size_t)t * 320 + 128;
        for (int e = 0; e < Dv; ++e) a += sr[e] * wr[e];
        hj[(size_t)row * 128 + t] = a;
    }
}

// ---------------------------------------------------------------------------
// Pair MLP with WMMA. One wave per 16-pair tile (fixed (b,i), 16 contiguous j).
// 8 waves / block (amortizes the 32KB weight staging across more tiles; CDNA5
// allows up to 320KB LDS per workgroup, we use ~81KB).
// GEMM1: 16x64 @ 64x128 (f16, f32 acc) -> +hi+hj+b1, ReLU ->
// GEMM2: 16x128 @ 128x64 -> +b2 + pair residual, mask, store.
// ---------------------------------------------------------------------------
#define PWAVES 8
__global__ void __launch_bounds__(32 * PWAVES)
pair_kernel(const float* __restrict__ pair_in,
            const unsigned char* __restrict__ mask,
            const float* __restrict__ pair_ln_w,
            const float* __restrict__ pair_ln_b,
            const _Float16* __restrict__ wp16,
            const _Float16* __restrict__ w216,
            const float* __restrict__ pmlp_b1,
            const float* __restrict__ pmlp_b2,
            const float* __restrict__ hi_g,
            const float* __restrict__ hj_g,
            float* __restrict__ pair_out) {
    __shared__ __align__(16) _Float16 s_wp[128 * 64]; // [h][p] : B1[k=p][n=h] = s_wp[h*64+k]
    __shared__ __align__(16) _Float16 s_w2[64 * 128]; // [n][k] : B2[k][n]    = s_w2[n*128+k]
    __shared__ float s_b1[128];
    __shared__ float s_b2[64];
    __shared__ float s_lnw[64];
    __shared__ float s_lnb[64];
    __shared__ __align__(16) float    s_r0[PWAVES][1024]; // fp32 tile / f16 hidden / fp32 out
    __shared__ __align__(16) _Float16 s_r1[PWAVES][1024]; // f16 pair_norm A tile

    int tid = threadIdx.x;
    // stage weights with b128 copies (1024 uint4 per array)
    {
        const uint4* gw = (const uint4*)wp16;
        const uint4* g2 = (const uint4*)w216;
        uint4* lw = (uint4*)s_wp;
        uint4* l2 = (uint4*)s_w2;
        for (int i = tid; i < 1024; i += 32 * PWAVES) { lw[i] = gw[i]; l2[i] = g2[i]; }
    }
    if (tid < 128) s_b1[tid] = pmlp_b1[tid];
    if (tid < 64) { s_b2[tid] = pmlp_b2[tid]; s_lnw[tid] = pair_ln_w[tid]; s_lnb[tid] = pair_ln_b[tid]; }
    __syncthreads();

    int wid  = tid >> 5;
    int lane = tid & 31;
    int tile = blockIdx.x * PWAVES + wid;   // tiles = B * L * (L/16) = 32768
    int jt    = tile & 31;                  // L/16 = 32
    int i_idx = (tile >> 5) & 511;
    int b     = tile >> 14;
    int j0    = jt << 4;

    float*    r0 = s_r0[wid];
    _Float16* r1 = s_r1[wid];

    size_t tbase = ((size_t)(b * Lv + i_idx) * Lv + j0) * Pv;

    // --- load pair tile: contiguous 1024 floats, coalesced float4 ---
    {
        const float4* src = (const float4*)(pair_in + tbase);
        float4* dst = (float4*)r0;
        for (int k = lane; k < 256; k += 32) dst[k] = src[k];
    }
    wavesync();

    // --- LayerNorm per row (2 lanes per row, shfl-combine) -> f16 into r1 ---
    {
        int rrow = lane & 15, half = lane >> 4;
        const float* rp = r0 + rrow * 64 + half * 32;
        float s = 0.f, sq = 0.f;
#pragma unroll
        for (int p = 0; p < 32; ++p) { float v = rp[p]; s += v; sq += v * v; }
        s  += __shfl_xor(s, 16, 32);
        sq += __shfl_xor(sq, 16, 32);
        float mean = s * (1.f / 64.f);
        float var  = sq * (1.f / 64.f) - mean * mean;
        float rstd = rsqrtf(var + 1e-5f);
        _Float16* op = r1 + rrow * 64 + half * 32;
#pragma unroll
        for (int p = 0; p < 32; ++p) {
            int pc = half * 32 + p;
            op[p] = (_Float16)((rp[p] - mean) * rstd * s_lnw[pc] + s_lnb[pc]);
        }
    }
    wavesync();

    int mrow = lane & 15;   // A-matrix row for this lane
    int klo  = lane >> 4;   // A-matrix K half selector
    int nloc = lane & 15;   // B/C column for this lane
    int kblk = lane >> 4;   // B-matrix K block selector
    int rowoff = (lane >> 4) * 8; // C-matrix row offset

    // --- GEMM1: A = pair_norm (16x64), B = Wp^T (64x128), acc f32 ---
    h16 aA[2];
#pragma unroll
    for (int c = 0; c < 2; ++c) {
        h8 lo = *(const h8*)(r1 + mrow * 64 + c * 32 + klo * 16);
        h8 hi = *(const h8*)(r1 + mrow * 64 + c * 32 + klo * 16 + 8);
        aA[c] = __builtin_shufflevector(lo, hi, 0, 1, 2, 3, 4, 5, 6, 7, 8, 9, 10, 11, 12, 13, 14, 15);
    }
    v8f acc[8] = {};
#pragma unroll
    for (int nt = 0; nt < 8; ++nt) {
#pragma unroll
        for (int c = 0; c < 2; ++c) {
            const h8* q = (const h8*)(s_wp + (nt * 16 + nloc) * 64 + c * 32 + kblk * 16);
            h16 bf = __builtin_shufflevector(q[0], q[1], 0, 1, 2, 3, 4, 5, 6, 7, 8, 9, 10, 11, 12, 13, 14, 15);
            acc[nt] = __builtin_amdgcn_wmma_f32_16x16x32_f16(false, aA[c], false, bf,
                                                             (short)0, acc[nt], false, false);
        }
    }

    // --- epilogue1: + hi + hj + b1, ReLU -> f16 hidden into r0 ---
    _Float16* hid = (_Float16*)r0;
    const float* hip = hi_g + (size_t)(b * Lv + i_idx) * 128;
#pragma unroll
    for (int nt = 0; nt < 8; ++nt) {
        int h = nt * 16 + nloc;
        float hv = hip[h] + s_b1[h];
#pragma unroll
        for (int r = 0; r < 8; ++r) {
            int m = r + rowoff;
            float hjv = hj_g[(size_t)(b * Lv + j0 + m) * 128 + h];
            float v = acc[nt][r] + hv + hjv;
            hid[m * 128 + h] = (_Float16)fmaxf(v, 0.f);
        }
    }
    wavesync();

    // --- GEMM2: A = hidden (16x128), B = W2^T (128x64) ---
    v8f acc2[4] = {};
#pragma unroll
    for (int c = 0; c < 4; ++c) {
        h8 lo = *(const h8*)(hid + mrow * 128 + c * 32 + klo * 16);
        h8 hi = *(const h8*)(hid + mrow * 128 + c * 32 + klo * 16 + 8);
        h16 a2 = __builtin_shufflevector(lo, hi, 0, 1, 2, 3, 4, 5, 6, 7, 8, 9, 10, 11, 12, 13, 14, 15);
#pragma unroll
        for (int nt = 0; nt < 4; ++nt) {
            const h8* q = (const h8*)(s_w2 + (nt * 16 + nloc) * 128 + c * 32 + kblk * 16);
            h16 bf = __builtin_shufflevector(q[0], q[1], 0, 1, 2, 3, 4, 5, 6, 7, 8, 9, 10, 11, 12, 13, 14, 15);
            acc2[nt] = __builtin_amdgcn_wmma_f32_16x16x32_f16(false, a2, false, bf,
                                                              (short)0, acc2[nt], false, false);
        }
    }
    wavesync();

    // --- epilogue2: stage fp32 result (+b2) into r0 for coalesced store ---
#pragma unroll
    for (int nt = 0; nt < 4; ++nt) {
        int n = nt * 16 + nloc;
        float bv = s_b2[n];
#pragma unroll
        for (int r = 0; r < 8; ++r) {
            int m = r + rowoff;
            r0[m * 64 + n] = acc2[nt][r] + bv;
        }
    }
    wavesync();

    // --- residual + mask, coalesced b128 load/store (mask uniform per row) ---
    float mi = mask[b * Lv + i_idx] ? 1.f : 0.f;
    {
        const float4* pin = (const float4*)(pair_in + tbase);
        float4* pout = (float4*)(pair_out + tbase);
        const float4* lres = (const float4*)r0;
        for (int q4 = lane; q4 < 256; q4 += 32) {
            int m = q4 >> 4; // 16 float4 per row
            float mm = mi * (mask[b * Lv + j0 + m] ? 1.f : 0.f);
            float4 a = pin[q4];
            float4 c = lres[q4];
            float4 o;
            o.x = (a.x + c.x) * mm;
            o.y = (a.y + c.y) * mm;
            o.z = (a.z + c.z) * mm;
            o.w = (a.w + c.w) * mm;
            pout[q4] = o;
        }
    }
}

// ---------------------------------------------------------------------------
extern "C" void kernel_launch(void* const* d_in, const int* in_sizes, int n_in,
                              void* d_out, int out_size, void* d_ws, size_t ws_size,
                              hipStream_t stream) {
    const float* residue_repr       = (const float*)d_in[0];
    const float* pair_repr          = (const float*)d_in[1];
    const unsigned char* mask       = (const unsigned char*)d_in[2];
    const float* ln1_w              = (const float*)d_in[3];
    const float* ln1_b              = (const float*)d_in[4];
    const float* in_proj_w          = (const float*)d_in[5];
    const float* in_proj_b          = (const float*)d_in[6];
    const float* out_proj_w         = (const float*)d_in[7];
    const float* out_proj_b         = (const float*)d_in[8];
    const float* ln2_w              = (const float*)d_in[9];
    const float* ln2_b              = (const float*)d_in[10];
    const float* ffn_w1             = (const float*)d_in[11];
    const float* ffn_b1             = (const float*)d_in[12];
    const float* ffn_w2             = (const float*)d_in[13];
    const float* ffn_b2             = (const float*)d_in[14];
    const float* pair_ln_w          = (const float*)d_in[15];
    const float* pair_ln_b          = (const float*)d_in[16];
    const float* pmlp_w1            = (const float*)d_in[17];
    const float* pmlp_b1            = (const float*)d_in[18];
    const float* pmlp_w2            = (const float*)d_in[19];
    const float* pmlp_b2            = (const float*)d_in[20];

    // workspace carve
    float* qkv = (float*)d_ws;                       // 1024*384 f32
    float* ctx = qkv + 1024 * 384;                   // 1024*128 f32
    float* hi  = ctx + 1024 * 128;                   // 1024*128 f32
    float* hj  = hi  + 1024 * 128;                   // 1024*128 f32
    _Float16* wp16 = (_Float16*)(hj + 1024 * 128);   // 8192 f16
    _Float16* w216 = wp16 + 8192;                    // 8192 f16

    float* res_out  = (float*)d_out;                 // B*L*D
    float* pair_out = res_out + Bv * Lv * Dv;        // B*L*L*P

    wprep_kernel<<<32, 256, 0, stream>>>(pmlp_w1, pmlp_w2, wp16, w216);
    ln_qkv_kernel<<<Bv * Lv, 128, 0, stream>>>(residue_repr, ln1_w, ln1_b,
                                               in_proj_w, in_proj_b, qkv);
    attn_kernel<<<Bv * Hv * Lv, 128, 0, stream>>>(qkv, mask, ctx);
    outffn_kernel<<<Bv * Lv, 256, 0, stream>>>(residue_repr, ctx,
                                               out_proj_w, out_proj_b,
                                               ln2_w, ln2_b,
                                               ffn_w1, ffn_b1, ffn_w2, ffn_b2,
                                               mask, res_out);
    hij_kernel<<<Bv * Lv, 256, 0, stream>>>(res_out, pmlp_w1, hi, hj);

    int tiles = Bv * Lv * (Lv / 16);                 // 32768
    pair_kernel<<<tiles / PWAVES, 32 * PWAVES, 0, stream>>>(
        pair_repr, mask, pair_ln_w, pair_ln_b, wp16, w216,
        pmlp_b1, pmlp_b2, hi, hj, pair_out);
}
